// CrossAttention_26388279066777
// MI455X (gfx1250) — matrix operand order repaired and checked by reading, not compiled, not run on hardware
//
#include <hip/hip_runtime.h>
#include <hip/hip_bf16.h>

typedef _Float16 f16;
typedef __attribute__((ext_vector_type(16))) _Float16 v16h;
typedef __attribute__((ext_vector_type(8)))  _Float16 v8h;
typedef __attribute__((ext_vector_type(8)))  float    v8f;

union AF { v16h v; v8h h[2]; };

__device__ __forceinline__ v8f wmma16(v16h a, v16h b, v8f c) {
  // v_wmma_f32_16x16x32_f16  (neg_a, A, neg_b, B, c_mod, C, reuse_a, reuse_b)
  return __builtin_amdgcn_wmma_f32_16x16x32_f16(false, a, false, b, (short)0, c, false, false);
}

static constexpr int D_ = 512;   // query dim / inner dim
static constexpr int N_ = 4096;  // sequence length
static constexpr int B_ = 2;     // batch
static constexpr int DH = 64;    // dim per head

// ---- CDNA5 async global->LDS (ASYNCcnt path), per CDNA5 ISA ----
// LDS flat aperture base has zero low-32 bits, so the low 32 bits of a flat
// pointer to __shared__ ARE the LDS byte offset.
__device__ __forceinline__ void async_ld_b128(f16* lds, const f16* g) {
  unsigned l = (unsigned)(uintptr_t)lds;
  asm volatile("global_load_async_to_lds_b128 %0, %1, off" :: "v"(l), "v"(g) : "memory");
}
__device__ __forceinline__ void wait_async0() {
  asm volatile("s_wait_asynccnt 0x0" ::: "memory");
}
__device__ __forceinline__ void wait_async4() {
  asm volatile("s_wait_asynccnt 0x4" ::: "memory");
}

// ---------------- conversion kernels ----------------
__global__ void cvt_x_k(const float* __restrict__ x, f16* __restrict__ xo, int n) {
  int i = blockIdx.x * blockDim.x + threadIdx.x;
  if (i < n) xo[i] = (f16)x[i];
}

// W is [512(in)][512(out)] f32; produce Wt [512(out)][512(in)] f16
__global__ void cvt_wt_k(const float* __restrict__ W, f16* __restrict__ Wt) {
  int i = blockIdx.x * blockDim.x + threadIdx.x;   // 262144 threads
  int o = i >> 9, in = i & 511;
  Wt[o * 512 + in] = (f16)W[in * 512 + o];
}

__device__ __forceinline__ void loadA(AF& a, const f16* arow, int k0) {
  a.h[0] = *(const v8h*)(arow + k0);          // K = k0 + hf*16 + 0..7
  a.h[1] = *(const v8h*)(arow + k0 + 8);      // K = k0 + hf*16 + 8..15
}

// ---------------- QKV projection: X[8192x512] @ W -> Q/K/V ----------------
// blockIdx.x = 16-row tile (512), blockIdx.y = matrix (0=Q,1=K,2=V)
// 8 waves/block, each wave a 16x64 slab (4 accumulator chains), k-loop software-pipelined
__global__ __launch_bounds__(256) void qkv_k(const f16* __restrict__ Xb,
    const f16* __restrict__ Wqt, const f16* __restrict__ Wkt, const f16* __restrict__ Wvt,
    f16* __restrict__ Qg, f16* __restrict__ Kg, f16* __restrict__ Vt)
{
  const int wave = threadIdx.x >> 5, lane = threadIdx.x & 31;
  const int M = lane & 15, hf = lane >> 4;
  const int mat = blockIdx.y;
  const f16* Wt = (mat == 0) ? Wqt : (mat == 1) ? Wkt : Wvt;
  const int row0 = blockIdx.x * 16;
  const int colBase = wave * 64;

  const f16* arow = Xb + (row0 + M) * 512 + hf * 16;
  const f16* brow = Wt + (colBase + M) * 512 + hf * 16;

  v8f acc[4] = {};
  AF a;  v16h b[4];
  loadA(a, arow, 0);
#pragma unroll
  for (int t = 0; t < 4; ++t) b[t] = *(const v16h*)(brow + t * 16 * 512);

#pragma unroll
  for (int k0 = 0; k0 < 512; k0 += 32) {
    AF an;  v16h bn[4];
    if (k0 + 32 < 512) {
      loadA(an, arow, k0 + 32);
#pragma unroll
      for (int t = 0; t < 4; ++t) bn[t] = *(const v16h*)(brow + t * 16 * 512 + k0 + 32);
    }
#pragma unroll
    for (int t = 0; t < 4; ++t) acc[t] = wmma16(a.v, b[t], acc[t]);
    if (k0 + 32 < 512) {
      a = an;
#pragma unroll
      for (int t = 0; t < 4; ++t) b[t] = bn[t];
    }
  }

  const float scale = (mat == 0) ? 0.125f : 1.0f;  // dh^-0.5 folded into Q
#pragma unroll
  for (int t = 0; t < 4; ++t) {
    const int c = colBase + t * 16 + M;
    const int head = c >> 6, d = c & 63;
#pragma unroll
    for (int r = 0; r < 8; ++r) {
      const int tok = row0 + r + hf * 8;
      const int b2 = tok >> 12, nTok = tok & 4095;
      const f16 v = (f16)(acc[t][r] * scale);
      if (mat == 2)      Vt[((b2 * 8 + head) * 64 + d) * 4096 + nTok] = v;   // V transposed
      else if (mat == 0) Qg[((b2 * 8 + head) * 4096 + nTok) * 64 + d] = v;
      else               Kg[((b2 * 8 + head) * 4096 + nTok) * 64 + d] = v;
    }
  }
}

// ---------------- fused flash attention ----------------
// blockIdx.y = b*8+h (16), blockIdx.x = 128-query block (32); wave owns 16 queries.
// K/V 64-key tiles double-buffered in LDS via async global->LDS loads.
__global__ __launch_bounds__(256) void flash_k(const f16* __restrict__ Qg,
    const f16* __restrict__ Kg, const f16* __restrict__ Vt, f16* __restrict__ Ob)
{
  __shared__ __align__(32) f16 Ks[2][64 * 64];    // [key][dh]
  __shared__ __align__(32) f16 Vs[2][64 * 64];    // [dh][key]  (V^T tile)
  __shared__ __align__(32) f16 Ps[8][16 * 64];    // per-wave P staging (C->A relayout)

  const int wave = threadIdx.x >> 5, lane = threadIdx.x & 31;
  const int M = lane & 15, hf = lane >> 4;
  const int bh = blockIdx.y;
  const int q0 = (blockIdx.x * 8 + wave) * 16;
  const f16* Qbh = Qg + bh * N_ * DH;
  const f16* Kbh = Kg + bh * N_ * DH;
  const f16* Vbh = Vt + bh * DH * N_;

  const int lrow = threadIdx.x >> 2, loff = (threadIdx.x & 3) * 16;

  // 4 async-load instructions per wave per tile (2 for K, 2 for V^T)
  auto issueTile = [&](int k0, int buf) {
    f16* lk = &Ks[buf][lrow * 64 + loff];
    const f16* gk = Kbh + (k0 + lrow) * 64 + loff;
    async_ld_b128(lk,     gk);
    async_ld_b128(lk + 8, gk + 8);
    f16* lv = &Vs[buf][lrow * 64 + loff];
    const f16* gv = Vbh + lrow * N_ + k0 + loff;
    async_ld_b128(lv,     gv);
    async_ld_b128(lv + 8, gv + 8);
  };

  issueTile(0, 0);
  issueTile(64, 1);

  // Q A-fragments for both 32-wide k-slices of dh=64, loaded once
  AF aQ[2];
  const f16* qrow = Qbh + (q0 + M) * 64 + hf * 16;
  aQ[0].h[0] = *(const v8h*)(qrow);
  aQ[0].h[1] = *(const v8h*)(qrow + 8);
  aQ[1].h[0] = *(const v8h*)(qrow + 32);
  aQ[1].h[1] = *(const v8h*)(qrow + 40);

  v8f o[4] = {};
  float m[8], l[8];
#pragma unroll
  for (int r = 0; r < 8; ++r) { m[r] = -3.0e38f; l[r] = 0.f; }

  for (int k0 = 0; k0 < N_; k0 += 64) {
    const int cur = (k0 >> 6) & 1;
    if (k0 + 64 >= N_) wait_async0();   // last tile: all our loads done
    else               wait_async4();   // tile i done, tile i+1 may be in flight
    __syncthreads();                    // tile i visible block-wide

    // S = Q K^T : 16 queries x 64 keys (4 C-tiles, 2 dh-slices each)
    v8f S[4] = {};
#pragma unroll
    for (int t = 0; t < 4; ++t)
#pragma unroll
      for (int ks = 0; ks < 2; ++ks) {
        v16h b = *(const v16h*)&Ks[cur][(t * 16 + M) * 64 + ks * 32 + hf * 16];
        S[t] = wmma16(aQ[ks].v, b, S[t]);
      }

    // online softmax (each row lives across the 16 lanes of a half-wave)
#pragma unroll
    for (int r = 0; r < 8; ++r) {
      float s = fmaxf(fmaxf(S[0][r], S[1][r]), fmaxf(S[2][r], S[3][r]));
      s = fmaxf(s, __shfl_xor(s, 1));
      s = fmaxf(s, __shfl_xor(s, 2));
      s = fmaxf(s, __shfl_xor(s, 4));
      s = fmaxf(s, __shfl_xor(s, 8));
      const float mn = fmaxf(m[r], s);
      const float corr = __expf(m[r] - mn);
      m[r] = mn;
      l[r] *= corr;
#pragma unroll
      for (int t = 0; t < 4; ++t) o[t][r] *= corr;
      float psum = 0.f;
#pragma unroll
      for (int t = 0; t < 4; ++t) {
        const float p = __expf(S[t][r] - mn);
        psum += p;
        Ps[wave][(r + hf * 8) * 64 + t * 16 + M] = (f16)p;   // row-major P[q][key]
      }
      l[r] += psum;   // per-lane partial; reduced once at the end
    }

    // reload P as A-fragments (same-wave LDS ops are in-order)
    AF aP[2];
    const f16* prow = &Ps[wave][M * 64 + hf * 16];
    aP[0].h[0] = *(const v8h*)(prow);
    aP[0].h[1] = *(const v8h*)(prow + 8);
    aP[1].h[0] = *(const v8h*)(prow + 32);
    aP[1].h[1] = *(const v8h*)(prow + 40);

    // O += P V : B-fragments straight out of the V^T tile
#pragma unroll
    for (int t = 0; t < 4; ++t)
#pragma unroll
      for (int ks = 0; ks < 2; ++ks) {
        v16h b = *(const v16h*)&Vs[cur][(t * 16 + M) * 64 + ks * 32 + hf * 16];
        o[t] = wmma16(aP[ks].v, b, o[t]);
      }

    __syncthreads();                    // everyone done reading buffer `cur`
    if (k0 + 128 < N_) issueTile(k0 + 128, cur);   // refill freed buffer
  }

#pragma unroll
  for (int r = 0; r < 8; ++r) {
    float s = l[r];
    s += __shfl_xor(s, 1);
    s += __shfl_xor(s, 2);
    s += __shfl_xor(s, 4);
    s += __shfl_xor(s, 8);
    l[r] = 1.0f / s;
  }
  const int b2 = bh >> 3, head = bh & 7;
#pragma unroll
  for (int t = 0; t < 4; ++t)
#pragma unroll
    for (int r = 0; r < 8; ++r) {
      const int tok = q0 + r + hf * 8;
      Ob[(b2 * N_ + tok) * 512 + head * 64 + t * 16 + M] = (f16)(o[t][r] * l[r]);
    }
}

// ---------------- output projection: Ob[8192x512] @ Wo + bo -> f32 ----------------
__global__ __launch_bounds__(256) void proj_k(const f16* __restrict__ Ob,
    const f16* __restrict__ Wot, const float* __restrict__ bo, float* __restrict__ out)
{
  const int wave = threadIdx.x >> 5, lane = threadIdx.x & 31;
  const int M = lane & 15, hf = lane >> 4;
  const int row0 = blockIdx.x * 16;
  const int colBase = wave * 64;

  const f16* arow = Ob + (row0 + M) * 512 + hf * 16;
  const f16* brow = Wot + (colBase + M) * 512 + hf * 16;

  v8f acc[4] = {};
  AF a;  v16h b[4];
  loadA(a, arow, 0);
#pragma unroll
  for (int t = 0; t < 4; ++t) b[t] = *(const v16h*)(brow + t * 16 * 512);

#pragma unroll
  for (int k0 = 0; k0 < 512; k0 += 32) {
    AF an;  v16h bn[4];
    if (k0 + 32 < 512) {
      loadA(an, arow, k0 + 32);
#pragma unroll
      for (int t = 0; t < 4; ++t) bn[t] = *(const v16h*)(brow + t * 16 * 512 + k0 + 32);
    }
#pragma unroll
    for (int t = 0; t < 4; ++t) acc[t] = wmma16(a.v, b[t], acc[t]);
    if (k0 + 32 < 512) {
      a = an;
#pragma unroll
      for (int t = 0; t < 4; ++t) b[t] = bn[t];
    }
  }

#pragma unroll
  for (int t = 0; t < 4; ++t) {
    const int c = colBase + t * 16 + M;
    const float bias = bo[c];
#pragma unroll
    for (int r = 0; r < 8; ++r) {
      const int tok = row0 + r + hf * 8;
      out[tok * 512 + c] = acc[t][r] + bias;
    }
  }
}

extern "C" void kernel_launch(void* const* d_in, const int* in_sizes, int n_in,
                              void* d_out, int out_size, void* d_ws, size_t ws_size,
                              hipStream_t stream)
{
  (void)in_sizes; (void)n_in; (void)out_size; (void)ws_size;
  const float* x  = (const float*)d_in[0];
  const float* Wq = (const float*)d_in[1];
  const float* Wk = (const float*)d_in[2];
  const float* Wv = (const float*)d_in[3];
  const float* Wo = (const float*)d_in[4];
  const float* bo = (const float*)d_in[5];
  float* out = (float*)d_out;

  const size_t NX = (size_t)B_ * N_ * D_;   // 4,194,304 elements
  char* ws = (char*)d_ws;
  f16* Xb  = (f16*)ws;  ws += NX * 2;
  f16* Wqt = (f16*)ws;  ws += 512 * 512 * 2;
  f16* Wkt = (f16*)ws;  ws += 512 * 512 * 2;
  f16* Wvt = (f16*)ws;  ws += 512 * 512 * 2;
  f16* Wot = (f16*)ws;  ws += 512 * 512 * 2;
  f16* Qg  = (f16*)ws;  ws += NX * 2;
  f16* Kg  = (f16*)ws;  ws += NX * 2;
  f16* Vt  = (f16*)ws;  ws += NX * 2;
  f16* Ob  = (f16*)ws;  ws += NX * 2;       // ~44 MB total

  cvt_x_k<<<(int)(NX / 256), 256, 0, stream>>>(x, Xb, (int)NX);
  cvt_wt_k<<<1024, 256, 0, stream>>>(Wq, Wqt);
  cvt_wt_k<<<1024, 256, 0, stream>>>(Wk, Wkt);
  cvt_wt_k<<<1024, 256, 0, stream>>>(Wv, Wvt);
  cvt_wt_k<<<1024, 256, 0, stream>>>(Wo, Wot);
  qkv_k<<<dim3(512, 3), 256, 0, stream>>>(Xb, Wqt, Wkt, Wvt, Qg, Kg, Vt);
  flash_k<<<dim3(32, 16), 256, 0, stream>>>(Qg, Kg, Vt, Ob);
  proj_k<<<512, 256, 0, stream>>>(Ob, Wot, bo, out);
}
